// MultiHeadAttention_33028298506753
// MI455X (gfx1250) — matrix operand order repaired and checked by reading, not compiled, not run on hardware
//
#include <hip/hip_runtime.h>
#include <hip/hip_bf16.h>

// MI455X / gfx1250, wave32. Mixed-precision (bf16 WMMA, fp32 accumulate)
// flash-attention MHA forward with TDM (tensor_load_to_lds) staging.
//
// Pipeline:
//   1) proj_qkv_kernel x3 : Q = xq@wq+bq -> bf16 [h][t][64]
//                           K = xk@wk+bk -> bf16 [h][t][64]
//                           V = xv@wv+bv -> bf16 [h][d][T]  (pre-transposed)
//   2) attn_kernel        : causal flash attention -> bf16 [t][768]
//                           K/V tiles staged to LDS by the Tensor Data Mover,
//                           double-buffered (TENSORcnt + workgroup barrier)
//   3) out_proj_kernel    : O@wo+bo -> fp32 d_out (TDM-staged bf16 A tile)

#define T_SEQ   4096
#define DMODEL  768
#define NHEADS  12
#define DK      64

typedef __attribute__((ext_vector_type(16))) __bf16         bf16x16;
typedef __attribute__((ext_vector_type(8)))  float          f32x8;
typedef __attribute__((ext_vector_type(4)))  float          f32x4;
typedef __attribute__((ext_vector_type(8)))  unsigned int   u32x8;
typedef __attribute__((ext_vector_type(4)))  unsigned int   u32x4;
typedef __attribute__((ext_vector_type(8)))  unsigned short u16x8;
typedef __attribute__((ext_vector_type(8)))  int            i32x8;
typedef __attribute__((ext_vector_type(4)))  int            i32x4;

// float -> bf16 (round to nearest even)
__device__ __forceinline__ unsigned short f2bf(float f) {
  unsigned int u = __builtin_bit_cast(unsigned int, f);
  u += 0x7FFFu + ((u >> 16) & 1u);
  return (unsigned short)(u >> 16);
}

// Load a 16-element bf16 fragment as two 16-byte chunks (b128 loads).
__device__ __forceinline__ bf16x16 ld_frag(const unsigned short* p, int e0, int e1) {
  u32x8 t;
  *(u32x4*)&t       = *(const u32x4*)(p + e0);
  *((u32x4*)&t + 1) = *(const u32x4*)(p + e1);
  return __builtin_bit_cast(bf16x16, t);
}

__device__ __forceinline__ f32x8 wmma_bf16(bf16x16 a, bf16x16 b, f32x8 c) {
  return __builtin_amdgcn_wmma_f32_16x16x32_bf16(false, a, false, b, (short)0, c,
                                                 false, false);
}

// ---------------------------------------------------------------------------
// TDM 2D tile load: global (bf16, row stride s0 elements) -> LDS at lds_off.
// D# built per CDNA5 ISA. w0pad carries group1
// word0: data_size=2B (bits17:16=1) plus pad_enable/pad_interval/pad_amount.
// This toolchain exposes the 6-arg builtin:
//   (u32x4 g0, i32x8 g1, i32x4 g2, i32x4 g3, i32x8 extra, i32 cpol)
// ---------------------------------------------------------------------------
__device__ __forceinline__ void tdm_load_2d(unsigned lds_off,
                                            const unsigned short* g,
                                            unsigned tile_d0, unsigned tile_d1,
                                            unsigned long long s0,
                                            unsigned w0pad) {
  unsigned long long ga = (unsigned long long)(const void*)g;
  const unsigned tensor_d0 = (unsigned)s0;     // row capacity (interior tiles)
  const unsigned tensor_d1 = 1u << 20;         // large: no y clamping
  u32x4 g0;
  g0[0] = 1u;                                   // count=1, user descriptor
  g0[1] = lds_off;                              // lds_addr (bytes)
  g0[2] = (unsigned)(ga & 0xFFFFFFFFu);         // global_addr[31:0]
  g0[3] = (unsigned)((ga >> 32) & 0x1FFFFFFu) | (2u << 30);  // type=2 (image)
  i32x8 g1;
  g1[0] = (int)w0pad;                                          // mask/size/pad
  g1[1] = (int)((tensor_d0 & 0xFFFFu) << 16);                  // dim0[15:0]
  g1[2] = (int)((tensor_d0 >> 16) | ((tensor_d1 & 0xFFFFu) << 16));
  g1[3] = (int)((tensor_d1 >> 16) | (tile_d0 << 16));          // tile_dim0
  g1[4] = (int)(tile_d1 & 0xFFFFu);                            // tile_dim1
  g1[5] = (int)(unsigned)(s0 & 0xFFFFFFFFu);                   // stride0 lo
  g1[6] = (int)((unsigned)(s0 >> 32) & 0xFFFFu);               // stride0 hi
  g1[7] = 0;
  i32x4 gz4 = {0, 0, 0, 0};                     // 2D: groups 2/3 unused
  i32x8 gz8 = {0, 0, 0, 0, 0, 0, 0, 0};
  __builtin_amdgcn_tensor_load_to_lds(g0, g1, gz4, gz4, gz8, 0);
}

// group1 word0 helpers: data_size=1 (2 bytes), pad_enable, interval, amount
#define TDM_W0(intv, amt) (0x00010000u | (1u << 20) | ((intv) << 22) | ((amt) << 25))

// ---------------------------------------------------------------------------
// GEMM: out = X(fp32)[4096x768] @ W(fp32)[768x768] + bias, output bf16.
// vt_mode==0: out[(col/64)][row][col%64]  (head-major, for Q and K)
// vt_mode==1: out[col][row]               ([h*64+d][T], transposed V)
// ---------------------------------------------------------------------------
__global__ __launch_bounds__(128) void proj_qkv_kernel(
    const float* __restrict__ X, const float* __restrict__ W,
    const float* __restrict__ bias, unsigned short* __restrict__ out,
    int vt_mode) {
  __shared__ __align__(16) unsigned short As[64][40];   // [m][k] bf16
  __shared__ __align__(16) unsigned short Bs[64][40];   // transposed [n][k]

  const int tid  = threadIdx.x;
  const int lane = tid & 31;
  const int w    = tid >> 5;
  const int m0   = blockIdx.x * 64;
  const int n0   = blockIdx.y * 64;

  f32x8 acc[4];
  for (int t = 0; t < 4; ++t)
    for (int i = 0; i < 8; ++i) acc[t][i] = 0.0f;

  const int arow = tid >> 1, ahalf = tid & 1;
  const int brow = tid >> 2, bq    = tid & 3;

  for (int k0 = 0; k0 < DMODEL; k0 += 32) {
    { // stage A (64x32) fp32 -> bf16
      const float* xp = X + (size_t)(m0 + arow) * DMODEL + k0 + ahalf * 16;
      f32x4 v0 = ((const f32x4*)xp)[0];
      f32x4 v1 = ((const f32x4*)xp)[1];
      f32x4 v2 = ((const f32x4*)xp)[2];
      f32x4 v3 = ((const f32x4*)xp)[3];
      u16x8 lo, hi;
      for (int i = 0; i < 4; ++i) {
        lo[i] = f2bf(v0[i]); lo[4 + i] = f2bf(v1[i]);
        hi[i] = f2bf(v2[i]); hi[4 + i] = f2bf(v3[i]);
      }
      *(u16x8*)&As[arow][ahalf * 16]     = lo;
      *(u16x8*)&As[arow][ahalf * 16 + 8] = hi;
    }
    { // stage B (32x64) fp32 -> bf16, transposed to [n][k]
      const float* wp = W + (size_t)(k0 + brow) * DMODEL + n0 + bq * 16;
      f32x4 v0 = ((const f32x4*)wp)[0];
      f32x4 v1 = ((const f32x4*)wp)[1];
      f32x4 v2 = ((const f32x4*)wp)[2];
      f32x4 v3 = ((const f32x4*)wp)[3];
      float vals[16];
      *(f32x4*)&vals[0]  = v0; *(f32x4*)&vals[4]  = v1;
      *(f32x4*)&vals[8]  = v2; *(f32x4*)&vals[12] = v3;
#pragma unroll
      for (int j = 0; j < 16; ++j) Bs[bq * 16 + j][brow] = f2bf(vals[j]);
    }
    __syncthreads();

    const int kb = (lane < 16) ? 0 : 8;
    const int bb = (lane < 16) ? 0 : 16;
    bf16x16 a = ld_frag(&As[16 * w + (lane & 15)][0], kb, kb + 16);
#pragma unroll
    for (int t = 0; t < 4; ++t) {
      bf16x16 b = ld_frag(&Bs[16 * t + (lane & 15)][0], kb, kb + 16);
      acc[t] = wmma_bf16(a, b, acc[t]);
    }
    __syncthreads();
  }

  const int rbase = (lane < 16) ? 0 : 8;
  if (vt_mode) {
#pragma unroll
    for (int t = 0; t < 4; ++t) {
      const int gc = n0 + 16 * t + (lane & 15);
      const float bv = bias[gc];
#pragma unroll
      for (int r = 0; r < 8; ++r) {
        const int gr = m0 + 16 * w + rbase + r;
        out[(size_t)gc * T_SEQ + gr] = f2bf(acc[t][r] + bv);
      }
    }
  } else {
#pragma unroll
    for (int t = 0; t < 4; ++t) {
      const int gc = n0 + 16 * t + (lane & 15);
      const float bv = bias[gc];
#pragma unroll
      for (int r = 0; r < 8; ++r) {
        const int gr = m0 + 16 * w + rbase + r;
        out[((size_t)(gc >> 6) * T_SEQ + gr) * DK + (gc & 63)] = f2bf(acc[t][r] + bv);
      }
    }
  }
}

// ---------------------------------------------------------------------------
// Causal flash attention. One block = 64 queries of one head (4 waves x 16).
// TDM double-buffers the 32-key K (32x64) and V (64x32) tiles into LDS with
// a 16B row pad; wave 0 issues the DMA and waits TENSORcnt, barrier publishes.
// ---------------------------------------------------------------------------
__global__ __launch_bounds__(128) void attn_kernel(
    const unsigned short* __restrict__ Qb, const unsigned short* __restrict__ Kb,
    const unsigned short* __restrict__ VT, unsigned short* __restrict__ Ob) {
  // TDM-padded pitches: K rows 64 elems + 8 pad = 72; V rows 32 + 8 = 40
  __shared__ __align__(16) unsigned short Kt[2][32][72];
  __shared__ __align__(16) unsigned short Vt[2][64][40];
  __shared__ __align__(16) unsigned short Ps[4][16][40];  // per-wave P tile

  const int tid     = threadIdx.x;
  const int lane    = tid & 31;
  const int w       = tid >> 5;
  const int h       = blockIdx.y;
  const int qb      = blockIdx.x * 64;
  const int qw      = qb + 16 * w;
  const int halfsel = (lane < 16) ? 0 : 1;
  const int kb      = halfsel ? 8 : 0;           // A-frag half select
  const int bb      = halfsel ? 16 : 0;          // B-frag half select

  const unsigned kt_off[2] = {(unsigned)(size_t)(void*)&Kt[0][0][0],
                              (unsigned)(size_t)(void*)&Kt[1][0][0]};
  const unsigned vt_off[2] = {(unsigned)(size_t)(void*)&Vt[0][0][0],
                              (unsigned)(size_t)(void*)&Vt[1][0][0]};
  const unsigned short* kbase = Kb + (size_t)h * T_SEQ * DK;
  const unsigned short* vbase = VT + (size_t)h * DK * T_SEQ;

  // Q fragments (A-layout): row = qw + lane%16, two 32-wide d windows.
  const unsigned short* qrowp =
      Qb + ((size_t)h * T_SEQ + qw + (lane & 15)) * DK;
  bf16x16 aQ0 = ld_frag(qrowp, kb, kb + 16);           // d 0..31
  bf16x16 aQ1 = ld_frag(qrowp, 32 + kb, 32 + kb + 16); // d 32..63

  f32x8 O[4];
  for (int t = 0; t < 4; ++t)
    for (int i = 0; i < 8; ++i) O[t][i] = 0.0f;
  float m_r[8], l_r[8];
  for (int r = 0; r < 8; ++r) { m_r[r] = -1e30f; l_r[r] = 0.0f; }

  const int nkb = (qb + 64) / 32;   // causal: keys 0 .. qb+63

  // prologue: DMA block 0 into buffer 0
  if (w == 0) {
    tdm_load_2d(kt_off[0], kbase, 64, 32, 64, TDM_W0(4u, 3u));      // K 32x64
    tdm_load_2d(vt_off[0], vbase, 32, 64, T_SEQ, TDM_W0(3u, 3u));   // V 64x32
    __builtin_amdgcn_s_wait_tensorcnt(0);
  }
  __syncthreads();

  for (int kblk = 0; kblk < nkb; ++kblk) {
    const int kk  = kblk * 32;
    const int cur = kblk & 1;
    // async DMA of the next block overlaps this block's compute
    if (w == 0 && kblk + 1 < nkb) {
      tdm_load_2d(kt_off[cur ^ 1], kbase + (size_t)(kk + 32) * DK, 64, 32, 64,
                  TDM_W0(4u, 3u));
      tdm_load_2d(vt_off[cur ^ 1], vbase + (kk + 32), 32, 64, T_SEQ,
                  TDM_W0(3u, 3u));
    }

    // ---- scores S = (Q K^T) / 8, two 16-key subtiles -------------------
    f32x8 S[2];
#pragma unroll
    for (int t = 0; t < 2; ++t) {
      const unsigned short* krowp = &Kt[cur][16 * t + (lane & 15)][0];
      bf16x16 b0 = ld_frag(krowp, kb, kb + 16);            // d 0..31
      bf16x16 b1 = ld_frag(krowp, 32 + kb, 32 + kb + 16);  // d 32..63
      f32x8 z; for (int i = 0; i < 8; ++i) z[i] = 0.0f;
      f32x8 s = wmma_bf16(aQ0, b0, z);
      S[t]    = wmma_bf16(aQ1, b1, s);
    }
    // scale + causal mask
#pragma unroll
    for (int t = 0; t < 2; ++t) {
      const int key = kk + 16 * t + (lane & 15);
#pragma unroll
      for (int r = 0; r < 8; ++r) {
        const int q = qw + r + 8 * halfsel;
        const float s = S[t][r] * 0.125f;
        S[t][r] = (key <= q) ? s : -1e30f;
      }
    }

    // ---- online softmax: per-row max/sum over 16 lanes of each half ----
    float cf[8];
#pragma unroll
    for (int r = 0; r < 8; ++r) {
      float mb = fmaxf(S[0][r], S[1][r]);
      for (int msk = 1; msk < 16; msk <<= 1)
        mb = fmaxf(mb, __shfl_xor(mb, msk, 32));
      const float mn = fmaxf(m_r[r], mb);
      cf[r] = __expf(m_r[r] - mn);
      m_r[r] = mn;
      const float p0 = __expf(S[0][r] - mn);
      const float p1 = __expf(S[1][r] - mn);
      Ps[w][r + 8 * halfsel][(lane & 15)]      = f2bf(p0);
      Ps[w][r + 8 * halfsel][16 + (lane & 15)] = f2bf(p1);
      float rs = p0 + p1;
      for (int msk = 1; msk < 16; msk <<= 1)
        rs += __shfl_xor(rs, msk, 32);
      l_r[r] = l_r[r] * cf[r] + rs;
    }
#pragma unroll
    for (int t = 0; t < 4; ++t)
#pragma unroll
      for (int r = 0; r < 8; ++r) O[t][r] *= cf[r];

    // wave-private LDS transpose (C-layout -> A-layout): wait for P stores
    asm volatile("s_wait_dscnt 0x0" ::: "memory");

    // ---- O += P(16x32) @ V(32x64) --------------------------------------
    bf16x16 aP = ld_frag(&Ps[w][lane & 15][0], kb, kb + 16);
#pragma unroll
    for (int td = 0; td < 4; ++td) {
      bf16x16 bV = ld_frag(&Vt[cur][16 * td + (lane & 15)][0], kb, kb + 16);
      O[td] = wmma_bf16(aP, bV, O[td]);
    }

    // retire next-block DMA, then publish to all waves
    if (w == 0 && kblk + 1 < nkb) __builtin_amdgcn_s_wait_tensorcnt(0);
    __syncthreads();
  }

  // ---- normalize by l and store bf16 token-major [t][h*64+d] -----------
#pragma unroll
  for (int r = 0; r < 8; ++r) {
    const float inv = (l_r[r] > 0.0f) ? (1.0f / l_r[r]) : 0.0f;
    const int q = qw + r + 8 * halfsel;
#pragma unroll
    for (int td = 0; td < 4; ++td) {
      const int d = 16 * td + (lane & 15);
      Ob[(size_t)q * DMODEL + h * DK + d] = f2bf(O[td][r] * inv);
    }
  }
}

// ---------------------------------------------------------------------------
// Output projection: out(fp32) = A(bf16)[4096x768] @ W(fp32)[768x768] + bias
// A tile staged by TDM (64 rows x 32 cols, stride 768, 16B row pad -> [64][40])
// ---------------------------------------------------------------------------
__global__ __launch_bounds__(128) void out_proj_kernel(
    const unsigned short* __restrict__ Ain, const float* __restrict__ W,
    const float* __restrict__ bias, float* __restrict__ out) {
  __shared__ __align__(16) unsigned short As[64][40];
  __shared__ __align__(16) unsigned short Bs[64][40];

  const int tid  = threadIdx.x;
  const int lane = tid & 31;
  const int w    = tid >> 5;
  const int m0   = blockIdx.x * 64;
  const int n0   = blockIdx.y * 64;
  const unsigned as_off = (unsigned)(size_t)(void*)&As[0][0];

  f32x8 acc[4];
  for (int t = 0; t < 4; ++t)
    for (int i = 0; i < 8; ++i) acc[t][i] = 0.0f;

  const int brow = tid >> 2, bq = tid & 3;

  for (int k0 = 0; k0 < DMODEL; k0 += 32) {
    // stage A via TDM: tile 64 rows x 32 elems, row stride DMODEL
    if (w == 0)
      tdm_load_2d(as_off, Ain + (size_t)m0 * DMODEL + k0, 32, 64, DMODEL,
                  TDM_W0(3u, 3u));
    { // stage B transposed fp32 -> bf16
      const float* wp = W + (size_t)(k0 + brow) * DMODEL + n0 + bq * 16;
      f32x4 v0 = ((const f32x4*)wp)[0];
      f32x4 v1 = ((const f32x4*)wp)[1];
      f32x4 v2 = ((const f32x4*)wp)[2];
      f32x4 v3 = ((const f32x4*)wp)[3];
      float vals[16];
      *(f32x4*)&vals[0]  = v0; *(f32x4*)&vals[4]  = v1;
      *(f32x4*)&vals[8]  = v2; *(f32x4*)&vals[12] = v3;
#pragma unroll
      for (int j = 0; j < 16; ++j) Bs[bq * 16 + j][brow] = f2bf(vals[j]);
    }
    if (w == 0) __builtin_amdgcn_s_wait_tensorcnt(0);
    __syncthreads();

    const int kb = (lane < 16) ? 0 : 8;
    const int bb = (lane < 16) ? 0 : 16;
    bf16x16 a = ld_frag(&As[16 * w + (lane & 15)][0], kb, kb + 16);
#pragma unroll
    for (int t = 0; t < 4; ++t) {
      bf16x16 b = ld_frag(&Bs[16 * t + (lane & 15)][0], kb, kb + 16);
      acc[t] = wmma_bf16(a, b, acc[t]);
    }
    __syncthreads();
  }

  const int rbase = (lane < 16) ? 0 : 8;
#pragma unroll
  for (int t = 0; t < 4; ++t) {
    const int gc = n0 + 16 * t + (lane & 15);
    const float bv = bias[gc];
#pragma unroll
    for (int r = 0; r < 8; ++r) {
      const int gr = m0 + 16 * w + rbase + r;
      out[(size_t)gr * DMODEL + gc] = acc[t][r] + bv;
    }
  }
}

// ---------------------------------------------------------------------------
extern "C" void kernel_launch(void* const* d_in, const int* in_sizes, int n_in,
                              void* d_out, int out_size, void* d_ws, size_t ws_size,
                              hipStream_t stream) {
  (void)in_sizes; (void)n_in; (void)out_size; (void)ws_size;
  const float* xq = (const float*)d_in[0];
  const float* xk = (const float*)d_in[1];
  const float* xv = (const float*)d_in[2];
  // d_in[3] = boolean causal mask: implemented analytically in attn_kernel
  const float* wq = (const float*)d_in[4];
  const float* bq = (const float*)d_in[5];
  const float* wk = (const float*)d_in[6];
  const float* bk = (const float*)d_in[7];
  const float* wv = (const float*)d_in[8];
  const float* bv = (const float*)d_in[9];
  const float* wo = (const float*)d_in[10];
  const float* bo = (const float*)d_in[11];
  float* out = (float*)d_out;

  const size_t plane = (size_t)T_SEQ * DMODEL;   // elements per bf16 plane
  unsigned short* Qb = (unsigned short*)d_ws;    // [h][t][64]
  unsigned short* Kb = Qb + plane;               // [h][t][64]
  unsigned short* VT = Kb + plane;               // [h*64+d][T]
  unsigned short* Ob = VT + plane;               // [t][768]

  dim3 gproj(T_SEQ / 64, DMODEL / 64);
  proj_qkv_kernel<<<gproj, 128, 0, stream>>>(xq, wq, bq, Qb, 0);
  proj_qkv_kernel<<<gproj, 128, 0, stream>>>(xk, wk, bk, Kb, 0);
  proj_qkv_kernel<<<gproj, 128, 0, stream>>>(xv, wv, bv, VT, 1);
  attn_kernel<<<dim3(T_SEQ / 64, NHEADS), 128, 0, stream>>>(Qb, Kb, VT, Ob);
  out_proj_kernel<<<gproj, 128, 0, stream>>>(Ob, wo, bo, out);
}
